// chem_transformer2D_38860864094550
// MI455X (gfx1250) — matrix operand order repaired and checked by reading, not compiled, not run on hardware
//
#include <hip/hip_runtime.h>
#include <hip/hip_bf16.h>

// ---------------- types ----------------
typedef __attribute__((ext_vector_type(16))) __bf16 v16bf;
typedef __attribute__((ext_vector_type(8)))  float  v8f;

union Frag16 { v16bf v; unsigned short u[16]; };

// hardware f32->bf16 conversion (v_cvt on gfx1250), reinterpreted to ushort storage
__device__ __forceinline__ unsigned short f2bf(float f) {
  union { __bf16 b; unsigned short u; } x;
  x.b = (__bf16)f;
  return x.u;
}

__device__ __forceinline__ v8f wmma_bf16(v16bf a, v16bf b, v8f c) {
  // (neg_a, A, neg_b, B, c_mod, C, reuse_a, reuse_b)
  return __builtin_amdgcn_wmma_f32_16x16x32_bf16(false, a, false, b, (short)0, c,
                                                 false, false);
}

// A fragment (16x32 bf16), source bf16 row-major, lda in elements.
// lanes0-15 = M, h=lane>>4; elems[0..15]=K h*16+0..15
__device__ __forceinline__ v16bf load_a(const unsigned short* A, int lda,
                                        int row0, int k0, int lane) {
  int m = lane & 15, h = (lane >> 4) & 1;
  const unsigned short* p = A + (size_t)(row0 + m) * lda + k0;
  Frag16 f;
#pragma unroll
  for (int j = 0; j < 16; ++j) f.u[j] = p[h * 16 + j];
  return f.v;
}

// B fragment (32x16 bf16) from column-major storage Bt[n][k] (ldb over k).
__device__ __forceinline__ v16bf load_bt(const unsigned short* Bt, int ldb,
                                         int n0, int k0, int lane) {
  int n = lane & 15, h = (lane >> 4) & 1;
  const unsigned short* p = Bt + (size_t)(n0 + n) * ldb + k0 + h * 16;
  Frag16 f;
#pragma unroll
  for (int j = 0; j < 16; ++j) f.u[j] = p[j];
  return f.v;
}

// ---------------- prep kernels ----------------
// bulk f32 -> bf16 conversion, 4 elements/thread
__global__ __launch_bounds__(256) void cvt_bf16_kernel(const float* __restrict__ in,
                                                       unsigned short* __restrict__ out) {
  int idx = blockIdx.x * 256 + threadIdx.x;
  float4 v = ((const float4*)in)[idx];
  ushort4 o;
  o.x = f2bf(v.x); o.y = f2bf(v.y); o.z = f2bf(v.z); o.w = f2bf(v.w);
  ((ushort4*)out)[idx] = o;
}

// W[k][n] f32 (256x256) -> Wt[n][k] bf16  (B-fragment-friendly)
__global__ void transpose_w_kernel(const float* __restrict__ W,
                                   unsigned short* __restrict__ Wt) {
  int k = blockIdx.x * 16 + threadIdx.x;
  int n = blockIdx.y * 16 + threadIdx.y;
  Wt[n * 256 + k] = f2bf(W[k * 256 + n]);
}

__global__ void prep_bn_kernel(const float* __restrict__ g, const float* __restrict__ be,
                               const float* __restrict__ mean, const float* __restrict__ var,
                               float* __restrict__ scale, float* __restrict__ shift) {
  int i = threadIdx.x;
  float s = g[i] * rsqrtf(var[i] + 1e-3f);
  scale[i] = s;
  shift[i] = be[i] - mean[i] * s;
}

// ---------------- fused two-layer MLP ----------------
// X = bf16 [32768][256] (rows 0..16383 seq1, 16384..32767 seq2). 8 waves, 32x256 tile.
// mode 0: out row-major bf16 [32768][256]; mode 1: out transposed [2][16][256][1024]
__global__ __launch_bounds__(256) void mlp2_kernel(
    const unsigned short* __restrict__ X,
    const unsigned short* __restrict__ W1t, const float* __restrict__ b1,
    const unsigned short* __restrict__ W2t, const float* __restrict__ b2,
    unsigned short* __restrict__ outRM, unsigned short* __restrict__ outVT,
    int mode) {
  __shared__ unsigned short hid[32 * 264];   // hidden tile, bf16, padded stride
  const int lane = threadIdx.x & 31;
  const int w    = threadIdx.x >> 5;
  const int wr   = w >> 2, wc = w & 3;
  const int row0 = blockIdx.x * 32;
  const int n = lane & 15, h = (lane >> 4) & 1;

  // ---- layer 1: relu(X @ W1 + b1) -> LDS ----
  v8f acc[4] = {};
  for (int k0 = 0; k0 < 256; k0 += 32) {
    v16bf a = load_a(X, 256, row0 + wr * 16, k0, lane);
#pragma unroll
    for (int j = 0; j < 4; ++j) {
      v16bf bfr = load_bt(W1t, 256, wc * 64 + j * 16, k0, lane);
      acc[j] = wmma_bf16(a, bfr, acc[j]);
    }
  }
#pragma unroll
  for (int j = 0; j < 4; ++j) {
    int col = wc * 64 + j * 16 + n;
    float bias = b1[col];
#pragma unroll
    for (int i = 0; i < 8; ++i) {
      int r = wr * 16 + h * 8 + i;
      float v = acc[j][i] + bias;
      hid[r * 264 + col] = f2bf(v > 0.f ? v : 0.f);
    }
  }
  __syncthreads();

  // ---- layer 2: hidden @ W2 + b2 (linear) ----
  v8f acc2[4] = {};
  for (int k0 = 0; k0 < 256; k0 += 32) {
    v16bf a = load_a(hid, 264, wr * 16, k0, lane);
#pragma unroll
    for (int j = 0; j < 4; ++j) {
      v16bf bfr = load_bt(W2t, 256, wc * 64 + j * 16, k0, lane);
      acc2[j] = wmma_bf16(a, bfr, acc2[j]);
    }
  }
#pragma unroll
  for (int j = 0; j < 4; ++j) {
    int col = wc * 64 + j * 16 + n;
    float bias = b2[col];
#pragma unroll
    for (int i = 0; i < 8; ++i) {
      int r = row0 + wr * 16 + h * 8 + i;
      unsigned short bv = f2bf(acc2[j][i] + bias);
      if (mode == 0) {
        outRM[(size_t)r * 256 + col] = bv;
      } else {
        int seq = r >> 14;
        int b   = (r >> 10) & 15;
        int l   = r & 1023;
        outVT[(((size_t)seq * 16 + b) * 256 + col) * 1024 + l] = bv;
      }
    }
  }
}

// ---------------- scores: (Q . K^T) * disto / 256 ----------------
// each wave: 16x32 strip (A fragment reused across 2 WMMAs).
// grid (8 col-groups, 64 row-tiles, 16 batches), block = 4 waves.
__global__ __launch_bounds__(128) void scores_kernel(
    const unsigned short* __restrict__ Q,   // [16*1024][256] bf16 row-major
    const unsigned short* __restrict__ Km,  // [16*1024][256] bf16 row-major (== Bt)
    const float* __restrict__ disto,        // [16][1024][1024]
    float* __restrict__ S, int trans_d) {
  const int lane = threadIdx.x & 31;
  const int w    = threadIdx.x >> 5;
  const int b    = blockIdx.z;
  const int i0   = blockIdx.y * 16;
  const int j0   = blockIdx.x * 128 + w * 32;
  const unsigned short* A  = Q  + (size_t)b * 1024 * 256;
  const unsigned short* Bt = Km + (size_t)b * 1024 * 256;
  v8f acc[2] = {};
  for (int k0 = 0; k0 < 256; k0 += 32) {
    v16bf a = load_a(A, 256, i0, k0, lane);
#pragma unroll
    for (int j = 0; j < 2; ++j) {
      v16bf bf = load_bt(Bt, 256, j0 + j * 16, k0, lane);
      acc[j] = wmma_bf16(a, bf, acc[j]);
    }
  }
  const int n = lane & 15, h = (lane >> 4) & 1;
  const float* D = disto + (size_t)b * 1024 * 1024;
  float* So = S + (size_t)b * 1024 * 1024;
#pragma unroll
  for (int j = 0; j < 2; ++j) {
    int c = j0 + j * 16 + n;
#pragma unroll
    for (int i = 0; i < 8; ++i) {
      int r = i0 + h * 8 + i;
      float d = trans_d ? D[(size_t)c * 1024 + r] : D[(size_t)r * 1024 + c];
      So[(size_t)r * 1024 + c] = acc[j][i] * d * (1.0f / 256.0f);
    }
  }
}

// ---------------- column softmax (over axis i), output transposed bf16 ----------------
__global__ __launch_bounds__(256) void colsoftmax_kernel(
    const float* __restrict__ S, unsigned short* __restrict__ Wt) {
  int b = blockIdx.z;
  int j = blockIdx.x * 256 + threadIdx.x;
  const float* So = S + (size_t)b * 1024 * 1024 + j;
  float mx = -1e30f;
  for (int i = 0; i < 1024; ++i) mx = fmaxf(mx, So[(size_t)i * 1024]);
  float sum = 0.f;
  for (int i = 0; i < 1024; ++i) sum += __expf(So[(size_t)i * 1024] - mx);
  float inv = 1.0f / sum;
  unsigned short* W = Wt + (size_t)b * 1024 * 1024 + (size_t)j * 1024;
  for (int i = 0; i < 1024; ++i)
    W[i] = f2bf(__expf(So[(size_t)i * 1024] - mx) * inv);
}

// ---------------- apply: AP[b][j][s] = sum_i W[b][j][i] * VT[b][s][i] ----------------
__global__ __launch_bounds__(256) void ap_kernel(
    const unsigned short* __restrict__ Wt,  // [16][1024][1024] bf16 (A, row-major over i)
    const unsigned short* __restrict__ VT,  // [16][256][1024]  bf16 (Bt)
    unsigned short* __restrict__ AP) {      // [16][1024][256] bf16
  const int lane = threadIdx.x & 31;
  const int w    = threadIdx.x >> 5;
  const int wr   = w >> 2, wc = w & 3;
  const int b    = blockIdx.z;
  const int row0 = blockIdx.x * 32;
  const unsigned short* A  = Wt + (size_t)b * 1024 * 1024;
  const unsigned short* Bt = VT + (size_t)b * 256 * 1024;
  v8f acc[4] = {};
  for (int k0 = 0; k0 < 1024; k0 += 32) {
    v16bf a = load_a(A, 1024, row0 + wr * 16, k0, lane);
#pragma unroll
    for (int j = 0; j < 4; ++j) {
      v16bf bf = load_bt(Bt, 1024, wc * 64 + j * 16, k0, lane);
      acc[j] = wmma_bf16(a, bf, acc[j]);
    }
  }
  const int n = lane & 15, h = (lane >> 4) & 1;
  unsigned short* O = AP + (size_t)b * 1024 * 256;
#pragma unroll
  for (int j = 0; j < 4; ++j) {
    int c = wc * 64 + j * 16 + n;
#pragma unroll
    for (int i = 0; i < 8; ++i)
      O[(size_t)(row0 + wr * 16 + h * 8 + i) * 256 + c] = f2bf(acc[j][i]);
  }
}

// ---------------- FC + ReLU + BatchNorm(inference) ----------------
__global__ __launch_bounds__(256) void fc_bn_kernel(
    const unsigned short* __restrict__ AP,  // [16384][256] bf16
    const unsigned short* __restrict__ Wt,  // [256][256] bf16 col-major
    const float* __restrict__ bias,
    const float* __restrict__ bnscale, const float* __restrict__ bnshift,
    float* __restrict__ out) {
  const int lane = threadIdx.x & 31;
  const int w    = threadIdx.x >> 5;
  const int wr   = w >> 2, wc = w & 3;
  const int row0 = blockIdx.x * 32;
  v8f acc[4] = {};
  for (int k0 = 0; k0 < 256; k0 += 32) {
    v16bf a = load_a(AP, 256, row0 + wr * 16, k0, lane);
#pragma unroll
    for (int j = 0; j < 4; ++j) {
      v16bf bf = load_bt(Wt, 256, wc * 64 + j * 16, k0, lane);
      acc[j] = wmma_bf16(a, bf, acc[j]);
    }
  }
  const int n = lane & 15, h = (lane >> 4) & 1;
#pragma unroll
  for (int j = 0; j < 4; ++j) {
    int c = wc * 64 + j * 16 + n;
    float bi = bias[c], sc = bnscale[c], sh = bnshift[c];
#pragma unroll
    for (int i = 0; i < 8; ++i) {
      int r = row0 + wr * 16 + h * 8 + i;
      float v = acc[j][i] + bi;
      v = v > 0.f ? v : 0.f;
      out[(size_t)r * 256 + c] = v * sc + sh;
    }
  }
}

// ---------------- launch ----------------
extern "C" void kernel_launch(void* const* d_in, const int* in_sizes, int n_in,
                              void* d_out, int out_size, void* d_ws, size_t ws_size,
                              hipStream_t stream) {
  const float* seq1  = (const float*)d_in[0];
  const float* seq2  = (const float*)d_in[1];
  const float* disto = (const float*)d_in[2];
  const float* Wsrc[8] = {
    (const float*)d_in[3],  (const float*)d_in[5],   // kW1, kW2
    (const float*)d_in[7],  (const float*)d_in[9],   // qW1, qW2
    (const float*)d_in[11], (const float*)d_in[13],  // vW1, vW2
    (const float*)d_in[15], (const float*)d_in[17],  // fc1W, fc2W
  };
  const float* kb1 = (const float*)d_in[4];
  const float* kb2 = (const float*)d_in[6];
  const float* qb1 = (const float*)d_in[8];
  const float* qb2 = (const float*)d_in[10];
  const float* vb1 = (const float*)d_in[12];
  const float* vb2 = (const float*)d_in[14];
  const float* fc1b = (const float*)d_in[16];
  const float* fc2b = (const float*)d_in[18];
  const float* bn_g = (const float*)d_in[19];
  const float* bn_b = (const float*)d_in[20];
  const float* bn_m = (const float*)d_in[21];
  const float* bn_v = (const float*)d_in[22];

  char* ws = (char*)d_ws;
  size_t off = 0;
  auto take = [&](size_t bytes) -> char* {
    char* p = ws + off;
    off = (off + bytes + 255) & ~(size_t)255;
    return p;
  };
  unsigned short* Wt[8];
  for (int i = 0; i < 8; ++i) Wt[i] = (unsigned short*)take(256 * 256 * 2);
  float* bnsc = (float*)take(256 * 4);
  float* bnsh = (float*)take(256 * 4);
  unsigned short* Xbf   = (unsigned short*)take((size_t)32768 * 256 * 2);
  unsigned short* Kbuf  = (unsigned short*)take((size_t)32768 * 256 * 2);
  unsigned short* Qbuf  = (unsigned short*)take((size_t)32768 * 256 * 2);
  unsigned short* VTbuf = (unsigned short*)take((size_t)2 * 16 * 256 * 1024 * 2);
  float*          S     = (float*)take((size_t)16 * 1024 * 1024 * 4);
  unsigned short* WTb   = (unsigned short*)take((size_t)16 * 1024 * 1024 * 2);
  unsigned short* AP    = (unsigned short*)take((size_t)16 * 1024 * 256 * 2);
  (void)ws_size; (void)in_sizes; (void)n_in; (void)out_size;

  const size_t half = (size_t)16384 * 256;

  // prep: weights -> bf16 col-major, BN fold, inputs -> bf16
  for (int i = 0; i < 8; ++i)
    transpose_w_kernel<<<dim3(16, 16), dim3(16, 16), 0, stream>>>(Wsrc[i], Wt[i]);
  prep_bn_kernel<<<1, 256, 0, stream>>>(bn_g, bn_b, bn_m, bn_v, bnsc, bnsh);
  cvt_bf16_kernel<<<4096, 256, 0, stream>>>(seq1, Xbf);
  cvt_bf16_kernel<<<4096, 256, 0, stream>>>(seq2, Xbf + half);

  // K/Q/V MLPs (seq1 rows 0..16383, seq2 rows 16384..32767)
  mlp2_kernel<<<1024, 256, 0, stream>>>(Xbf, Wt[0], kb1, Wt[1], kb2, Kbuf, VTbuf, 0);
  mlp2_kernel<<<1024, 256, 0, stream>>>(Xbf, Wt[2], qb1, Wt[3], qb2, Qbuf, VTbuf, 0);
  mlp2_kernel<<<1024, 256, 0, stream>>>(Xbf, Wt[4], vb1, Wt[5], vb2, Kbuf, VTbuf, 1);

  // path 1: scores1 = q1.k2^T * disto/256 ; softmax over l ; ap1 = w1^T v1 ; fc1+bn
  scores_kernel<<<dim3(8, 64, 16), 128, 0, stream>>>(Qbuf, Kbuf + half, disto, S, 0);
  colsoftmax_kernel<<<dim3(4, 1, 16), 256, 0, stream>>>(S, WTb);
  ap_kernel<<<dim3(32, 1, 16), 256, 0, stream>>>(WTb, VTbuf, AP);
  fc_bn_kernel<<<512, 256, 0, stream>>>(AP, Wt[6], fc1b, bnsc, bnsh, (float*)d_out);

  // path 2: scores2 = q2.k1^T * disto^T/256 ; softmax over m ; ap2 = w2^T v2 ; fc2+bn
  scores_kernel<<<dim3(8, 64, 16), 128, 0, stream>>>(Qbuf + half, Kbuf, disto, S, 1);
  colsoftmax_kernel<<<dim3(4, 1, 16), 256, 0, stream>>>(S, WTb);
  ap_kernel<<<dim3(32, 1, 16), 256, 0, stream>>>(WTb, VTbuf + (size_t)16 * 256 * 1024, AP);
  fc_bn_kernel<<<512, 256, 0, stream>>>(AP, Wt[7], fc2b, bnsc, bnsh,
                                        (float*)d_out + (size_t)16 * 1024 * 256);
}
